// SPLoss_23682449670255
// MI455X (gfx1250) — matrix operand order repaired and checked by reading, not compiled, not run on hardware
//
#include <hip/hip_runtime.h>
#include <hip/hip_bf16.h>
#include <math.h>

typedef __attribute__((ext_vector_type(16))) _Float16 v16h;
typedef __attribute__((ext_vector_type(8)))  float    v8f;

#define BATCH 2
#define CHN   256
#define HW    4096
#define NPOS  (BATCH*HW)        // 8192 positions
#define LDT   72                // padded LDS row stride in halfs (144 B, 16B-aligned)

// workspace layout (bytes)
#define ACC_OFF   0ull
#define AT_OFF    256ull
#define TEN_BYTES ((size_t)BATCH*HW*CHN*2)   // 4 MiB per f16 tensor
#define BT_OFF    (AT_OFF + TEN_BYTES)
#define CT_OFF    (BT_OFF + TEN_BYTES)

union Frag16 { v16h h; uint4 u[2]; };

// ---------------------------------------------------------------- init
__global__ void init_acc_kernel(float* __restrict__ acc) {
    if (threadIdx.x < 4) acc[threadIdx.x] = 0.f;
}

// ---------------------------------------------------------------- kp loss
__global__ __launch_bounds__(256)
void kp_kernel(const float* __restrict__ sv, const float* __restrict__ si,
               const float* __restrict__ sf, float* __restrict__ acc) {
    __shared__ float red[256];
    const int tid = threadIdx.x;
    float s = 0.f;
    for (int i = blockIdx.x * blockDim.x + tid; i < NPOS; i += gridDim.x * blockDim.x)
        s += fabsf(sf[i] - fmaxf(sv[i], si[i]));
    red[tid] = s; __syncthreads();
    for (int off = 128; off; off >>= 1) {
        if (tid < off) red[tid] += red[tid + off];
        __syncthreads();
    }
    if (tid == 0) atomicAdd(&acc[0], red[0]);
}

// ------------------------------------------- normalize + desc loss + transpose
// One thread per pixel position; channel reads are coalesced across threads.
// Writes unit-norm descriptors as f16 in [pos, C] layout for the WMMA GEMM.
__global__ __launch_bounds__(256)
void prep_kernel(const float* __restrict__ dv, const float* __restrict__ di,
                 const float* __restrict__ df,
                 _Float16* __restrict__ aT, _Float16* __restrict__ bT,
                 _Float16* __restrict__ cT, float* __restrict__ acc) {
    __shared__ float red[256];
    const int tid = threadIdx.x;
    const int p   = blockIdx.x * 256 + tid;          // 0..8191
    const int b   = p >> 12;                         // /4096
    const int hw  = p & 4095;
    const size_t base = (size_t)b * CHN * HW + hw;

    float sa = 0.f, sb = 0.f, sc = 0.f, desc = 0.f;
    for (int c = 0; c < CHN; ++c) {
        const size_t idx = base + (size_t)c * HW;
        const float v = dv[idx], w = di[idx], f = df[idx];
        sa += v * v; sb += w * w; sc += f * f;
        desc += fabsf(f - (v + w) * 0.5f);
    }
    const float ra = 1.f / fmaxf(sqrtf(sa), 1e-12f);
    const float rb = 1.f / fmaxf(sqrtf(sb), 1e-12f);
    const float rc = 1.f / fmaxf(sqrtf(sc), 1e-12f);

    const size_t obase = (size_t)p * CHN;
    for (int c = 0; c < CHN; ++c) {
        const size_t idx = base + (size_t)c * HW;
        aT[obase + c] = (_Float16)(dv[idx] * ra);
        bT[obase + c] = (_Float16)(di[idx] * rb);
        cT[obase + c] = (_Float16)(df[idx] * rc);
    }

    red[tid] = desc; __syncthreads();
    for (int off = 128; off; off >>= 1) {
        if (tid < off) red[tid] += red[tid + off];
        __syncthreads();
    }
    if (tid == 0) atomicAdd(&acc[1], red[0]);
}

// ------------------------------------------- fused dual-GEMM + L1 reduction
// Block tile: 128 rows (a) x 128 cols (b AND c), K=256 blocked by 64 in LDS.
// 8 waves in a 2x4 grid; each wave owns a 64x32 patch = 4x2 WMMA subtiles for
// BOTH products -> 16 WMMAs per 16 ds_load_b128 per K=32 step.
__global__ __launch_bounds__(256)
void corr_kernel(const _Float16* __restrict__ aT, const _Float16* __restrict__ bT,
                 const _Float16* __restrict__ cT, float* __restrict__ acc) {
    __shared__ _Float16 As[128 * LDT];
    __shared__ _Float16 Bs[128 * LDT];
    __shared__ _Float16 Cs[128 * LDT];
    __shared__ float wsum[8];

    const int tid     = threadIdx.x;
    const int lane    = tid & 31;
    const int wave    = tid >> 5;
    const int batch   = blockIdx.z;
    const int rowBase = blockIdx.x * 128;
    const int colBase = blockIdx.y * 128;

    const uint4* a4 = (const uint4*)(aT + (size_t)batch * HW * CHN);
    const uint4* b4 = (const uint4*)(bT + (size_t)batch * HW * CHN);
    const uint4* c4 = (const uint4*)(cT + (size_t)batch * HW * CHN);

    const int wRow = (wave & 1) * 64;     // 2 wave-rows of 64
    const int wCol = (wave >> 1) * 32;    // 4 wave-cols of 32
    const int lm   = lane & 15;
    const int hiA  = (lane >> 4) * 16;    // A frag: lanes 16..31 start at K+16
    const int hiB  = (lane >> 4) * 16;    // B frag: lanes 16..31 start at K+16

    v8f AB[4][2] = {};                    // a x b
    v8f AC[4][2] = {};                    // a x c

    for (int kb = 0; kb < CHN; kb += 64) {
        // stage A, B, C tiles: 128 rows x 64 halfs each (8 x uint4 per row)
        #pragma unroll
        for (int it = 0; it < 4; ++it) {
            const int idx = tid + it * 256;         // 0..1023
            const int r = idx >> 3, j = idx & 7;
            const size_t ksrc = (size_t)(kb >> 3) + j;
            ((uint4*)&As[r * LDT])[j] = a4[(size_t)(rowBase + r) * 32 + ksrc];
            ((uint4*)&Bs[r * LDT])[j] = b4[(size_t)(colBase + r) * 32 + ksrc];
            ((uint4*)&Cs[r * LDT])[j] = c4[(size_t)(colBase + r) * 32 + ksrc];
        }
        __syncthreads();

        #pragma unroll
        for (int ks = 0; ks < 2; ++ks) {
            const int kA = ks * 32 + hiA;
            const int kB = ks * 32 + hiB;
            Frag16 a[4], fb[2], fc[2];
            #pragma unroll
            for (int i = 0; i < 4; ++i) {
                const uint4* p = (const uint4*)&As[(wRow + i * 16 + lm) * LDT + kA];
                a[i].u[0] = p[0]; a[i].u[1] = p[1];     // 16 contiguous K halfs
            }
            #pragma unroll
            for (int j = 0; j < 2; ++j) {
                const uint4* pb = (const uint4*)&Bs[(wCol + j * 16 + lm) * LDT + kB];
                fb[j].u[0] = pb[0]; fb[j].u[1] = pb[1]; // 16 contiguous K halfs
                const uint4* pc = (const uint4*)&Cs[(wCol + j * 16 + lm) * LDT + kB];
                fc[j].u[0] = pc[0]; fc[j].u[1] = pc[1];
            }
            #pragma unroll
            for (int i = 0; i < 4; ++i) {
                #pragma unroll
                for (int j = 0; j < 2; ++j) {
                    AB[i][j] = __builtin_amdgcn_wmma_f32_16x16x32_f16(
                        false, a[i].h, false, fb[j].h, (short)0, AB[i][j], false, false);
                    AC[i][j] = __builtin_amdgcn_wmma_f32_16x16x32_f16(
                        false, a[i].h, false, fc[j].h, (short)0, AC[i][j], false, false);
                }
            }
        }
        __syncthreads();
    }

    // |sim_ac - sim_ab| over this block's tile, reduced to one atomic
    float s = 0.f;
    #pragma unroll
    for (int i = 0; i < 4; ++i)
        #pragma unroll
        for (int j = 0; j < 2; ++j)
            #pragma unroll
            for (int e = 0; e < 8; ++e)
                s += fabsf(AC[i][j][e] - AB[i][j][e]);

    #pragma unroll
    for (int off = 16; off; off >>= 1) s += __shfl_xor(s, off, 32);
    if (lane == 0) wsum[wave] = s;
    __syncthreads();
    if (tid == 0) {
        float t = 0.f;
        #pragma unroll
        for (int w = 0; w < 8; ++w) t += wsum[w];
        atomicAdd(&acc[2], t);
    }
}

// ---------------------------------------------------------------- finalize
__global__ void finalize_kernel(const float* __restrict__ acc, float* __restrict__ out) {
    out[0] = acc[0] * (1.f / 8192.f)            // kp:    2*1*64*64
           + acc[1] * (1.f / 2097152.f)         // desc:  2*256*64*64
           + acc[2] * (1.f / 33554432.f);       // match: 2*4096*4096
}

extern "C" void kernel_launch(void* const* d_in, const int* in_sizes, int n_in,
                              void* d_out, int out_size, void* d_ws, size_t ws_size,
                              hipStream_t stream) {
    const float* sv = (const float*)d_in[0];
    const float* si = (const float*)d_in[1];
    const float* sf = (const float*)d_in[2];
    const float* dv = (const float*)d_in[3];
    const float* di = (const float*)d_in[4];
    const float* df = (const float*)d_in[5];

    float*    acc = (float*)((char*)d_ws + ACC_OFF);
    _Float16* aT  = (_Float16*)((char*)d_ws + AT_OFF);
    _Float16* bT  = (_Float16*)((char*)d_ws + BT_OFF);
    _Float16* cT  = (_Float16*)((char*)d_ws + CT_OFF);

    init_acc_kernel<<<1, 4, 0, stream>>>(acc);
    kp_kernel<<<8, 256, 0, stream>>>(sv, si, sf, acc);
    prep_kernel<<<NPOS / 256, 256, 0, stream>>>(dv, di, df, aT, bT, cT, acc);

    dim3 grid(HW / 128, HW / 128, BATCH);   // 32 x 32 x 2 blocks
    corr_kernel<<<grid, 256, 0, stream>>>(aT, bT, cT, acc);

    finalize_kernel<<<1, 1, 0, stream>>>(acc, (float*)d_out);
}
